// DenseSAKELayer_8031588844293
// MI455X (gfx1250) — matrix operand order repaired and checked by reading, not compiled, not run on hardware
//
#include <hip/hip_runtime.h>
#include <math.h>
#include <stdint.h>

#define BATCH 2
#define NN 256
#define FF 64
#define HH 64
#define NHEADS 4
#define CC 256
#define EPSV 1e-5f
#define INFV 1e5f
#define ZS 66   // padded LDS row stride in halves (breaks bank conflicts)

typedef __attribute__((ext_vector_type(16))) _Float16 v16h;
typedef __attribute__((ext_vector_type(8)))  float    v8f;
typedef __attribute__((ext_vector_type(4)))  unsigned int u32x4;
typedef __attribute__((ext_vector_type(8)))  int      i32x8;
typedef __attribute__((ext_vector_type(4)))  int      i32x4;

__device__ __forceinline__ float silu_f(float x) { return x / (1.0f + __expf(-x)); }

// ---------------------------------------------------------------------------
// Kernel 1: per-node partial products of edge layer-1.
//   A[node,k]  = h[node,:] @ edge_w1[0:64, k]
//   Bv[node,k] = h[node,:] @ edge_w1[64:128, k] + edge_b1[k]
// ---------------------------------------------------------------------------
__global__ __launch_bounds__(64) void k_prep(
    const float* __restrict__ h, const float* __restrict__ ew1,
    const float* __restrict__ eb1,
    float* __restrict__ Anode, float* __restrict__ Bnode)
{
    __shared__ float s_h[FF];
    int node = blockIdx.x;
    int t = threadIdx.x;
    s_h[t] = h[(size_t)node * FF + t];
    __syncthreads();
    float a = 0.0f, bb = eb1[t];
    #pragma unroll 4
    for (int f = 0; f < FF; ++f) {
        float hv = s_h[f];
        a  += hv * ew1[f * HH + t];
        bb += hv * ew1[(FF + f) * HH + t];
    }
    Anode[(size_t)node * HH + t] = a;
    Bnode[(size_t)node * HH + t] = bb;
}

// ---------------------------------------------------------------------------
// Kernel 2: edge MLP layer 2 via WMMA + semantic logits.
// One block per (b,i). M = 256 j-rows, K = 64, N = 64.
// Wave w owns n-tile (w&3); B fragments loaded once, reused for 8 m-tiles.
// ---------------------------------------------------------------------------
__global__ __launch_bounds__(256) void k_edge(
    const float* __restrict__ x, const float* __restrict__ Anode,
    const float* __restrict__ Bnode, const float* __restrict__ ew1,
    const float* __restrict__ ew2,  const float* __restrict__ eb2,
    const float* __restrict__ semw, const float* __restrict__ semb,
    _Float16* __restrict__ he, float* __restrict__ semo)
{
    __shared__ _Float16 s_z[NN * ZS];     // z1 (layer-1 silu), later h_e_mtx
    __shared__ _Float16 s_w2t[HH * ZS];   // edge_w2 transposed: [n][k]
    __shared__ float    s_dist[NN];

    int bi = blockIdx.x;
    int b  = bi >> 8;
    int i  = bi & 255;
    int t  = threadIdx.x;

    // stage edge_w2 transposed as f16
    for (int idx = t; idx < HH * HH; idx += 256) {
        int kk = idx >> 6, n = idx & 63;
        s_w2t[n * ZS + kk] = (_Float16)ew2[idx];
    }
    // distance row for this i (thread t = j)
    {
        int j = t;
        float dx = x[(b * NN + j) * 3 + 0] - x[(b * NN + i) * 3 + 0];
        float dy = x[(b * NN + j) * 3 + 1] - x[(b * NN + i) * 3 + 1];
        float dz = x[(b * NN + j) * 3 + 2] - x[(b * NN + i) * 3 + 2];
        s_dist[j] = sqrtf(dx * dx + dy * dy + dz * dz + EPSV);
    }
    __syncthreads();

    // layer 1 (factored) + SiLU -> f16 LDS
    for (int idx = t; idx < NN * HH; idx += 256) {
        int j = idx >> 6, kk = idx & 63;
        float pre = Anode[(b * NN + j) * HH + kk]
                  + Bnode[(b * NN + i) * HH + kk]
                  + s_dist[j] * ew1[128 * HH + kk];
        s_z[j * ZS + kk] = (_Float16)silu_f(pre);
    }
    __syncthreads();

    // ---- WMMA GEMM: 64 output tiles (16 m-tiles x 4 n-tiles), 2 K-steps ----
    int wave = t >> 5, lane = t & 31;
    int mrow = lane & 15, msel = lane >> 4;
    int nt = wave & 3;        // fixed n-tile for this wave
    int m0 = wave >> 2;       // 0 -> even m-tiles, 1 -> odd m-tiles

    // B tiles (32x16 f16): lane holds column n = lane&15; lanes<16 K 0-7 & 16-23, lanes>=16 K 8-15 & 24-31.
    union F16x16 { v16h v; uint32_t u[8]; };
    F16x16 bm[2];
    #pragma unroll
    for (int s = 0; s < 2; ++s) {
        const uint32_t* wr =
            (const uint32_t*)(&s_w2t[(nt * 16 + mrow) * ZS + 32 * s + msel * 8]);
        #pragma unroll
        for (int q = 0; q < 8; ++q) bm[s].u[q] = wr[q < 4 ? q : q + 4];
    }

    v8f accs[8];
    #pragma unroll
    for (int r = 0; r < 8; ++r) {
        int mt = m0 + 2 * r;
        v8f c;
        #pragma unroll
        for (int q = 0; q < 8; ++q) c[q] = 0.0f;

        #pragma unroll
        for (int s = 0; s < 2; ++s) {
            // A tile (16x32 f16): lane m = lane&15; lanes<16 hold K 0-7 & 16-23,
            // lanes>=16 hold K 8-15 & 24-31 (ISA 7.12.2 16-bit A layout).
            F16x16 a;
            const uint32_t* zr =
                (const uint32_t*)(&s_z[(mt * 16 + mrow) * ZS + 32 * s + msel * 8]);
            a.u[0] = zr[0]; a.u[1] = zr[1]; a.u[2] = zr[2]; a.u[3] = zr[3];
            a.u[4] = zr[8]; a.u[5] = zr[9]; a.u[6] = zr[10]; a.u[7] = zr[11];

            c = __builtin_amdgcn_wmma_f32_16x16x32_f16(
                    false, a.v, false, bm[s].v, (short)0, c, false, false);
        }
        accs[r] = c;
    }
    __syncthreads();  // all waves done reading s_z before overwrite

    // epilogue: bias + SiLU -> h_e_mtx, into LDS (f16) and global ws (f16)
    #pragma unroll
    for (int r = 0; r < 8; ++r) {
        int mt = m0 + 2 * r;
        int n = nt * 16 + mrow;
        float bias = eb2[n];
        #pragma unroll
        for (int vg = 0; vg < 8; ++vg) {
            int j = mt * 16 + vg + 8 * msel;   // C/D layout: vgpr g = row g + 8*msel
            float val = silu_f(accs[r][vg] + bias);
            _Float16 hv = (_Float16)val;
            s_z[j * ZS + n] = hv;
            he[((size_t)bi * NN + j) * HH + n] = hv;
        }
    }
    __syncthreads();

    // semantic logits: thread t = j, 4 heads, LeakyReLU(0.2)
    {
        int j = t;
        float a0 = semb[0], a1 = semb[1], a2 = semb[2], a3 = semb[3];
        #pragma unroll 4
        for (int kk = 0; kk < HH; ++kk) {
            float hv = (float)s_z[j * ZS + kk];
            a0 += hv * semw[kk * 4 + 0];
            a1 += hv * semw[kk * 4 + 1];
            a2 += hv * semw[kk * 4 + 2];
            a3 += hv * semw[kk * 4 + 3];
        }
        float* o = &semo[((size_t)bi * NN + j) * 4];
        o[0] = a0 > 0.0f ? a0 : 0.2f * a0;
        o[1] = a1 > 0.0f ? a1 : 0.2f * a1;
        o[2] = a2 > 0.0f ? a2 : 0.2f * a2;
        o[3] = a3 > 0.0f ? a3 : 0.2f * a3;
    }
}

// ---------------------------------------------------------------------------
// block-wide reductions (block size 256, all threads active)
// ---------------------------------------------------------------------------
__device__ __forceinline__ float blk_max256(float v, float* red) {
    int t = threadIdx.x;
    red[t] = v; __syncthreads();
    for (int s = 128; s > 0; s >>= 1) {
        if (t < s) red[t] = fmaxf(red[t], red[t + s]);
        __syncthreads();
    }
    float r = red[0]; __syncthreads();
    return r;
}
__device__ __forceinline__ float blk_sum256(float v, float* red) {
    int t = threadIdx.x;
    red[t] = v; __syncthreads();
    for (int s = 128; s > 0; s >>= 1) {
        if (t < s) red[t] += red[t + s];
        __syncthreads();
    }
    float r = red[0]; __syncthreads();
    return r;
}

// ---------------------------------------------------------------------------
// Kernel 3: euclid softmax * semantic softmax -> combined softmax (over j)
// ---------------------------------------------------------------------------
__global__ __launch_bounds__(256) void k_att(
    const float* __restrict__ x, const float* __restrict__ semo,
    const float* __restrict__ lgamma, float* __restrict__ atto)
{
    __shared__ float red[256];
    int bi = blockIdx.x;
    int b  = bi >> 8;
    int i  = bi & 255;
    int j  = threadIdx.x;

    float dx = x[(b * NN + j) * 3 + 0] - x[(b * NN + i) * 3 + 0];
    float dy = x[(b * NN + j) * 3 + 1] - x[(b * NN + i) * 3 + 1];
    float dz = x[(b * NN + j) * 3 + 2] - x[(b * NN + i) * 3 + 2];
    float dist = sqrtf(dx * dx + dy * dy + dz * dz + EPSV);
    float mask = (j == i) ? INFV : 0.0f;
    const float* sp = &semo[((size_t)bi * NN + j) * 4];

    for (int head = 0; head < NHEADS; ++head) {
        float g  = __expf(lgamma[head]);
        float el = -(dist + mask) * g;
        float m1 = blk_max256(el, red);
        float e1 = __expf(el - m1);
        float eu = e1 / blk_sum256(e1, red);

        float sl = sp[head] - mask;
        float m2 = blk_max256(sl, red);
        float e2 = __expf(sl - m2);
        float sa = e2 / blk_sum256(e2, red);

        float p  = eu * sa;
        float m3 = blk_max256(p, red);
        float e3 = __expf(p - m3);
        atto[((size_t)bi * NN + j) * 4 + head] = e3 / blk_sum256(e3, red);
    }
}

// ---------------------------------------------------------------------------
// Kernel 4: fused aggregation over j: h_agg, comb_norm, delta_v.
// h_e slab (256x64 f16 = 32KB) is DMA'd into LDS by the Tensor Data Mover.
// thread c in [0,256): hh = c>>2 (edge feature), head = c&3
// ---------------------------------------------------------------------------
__global__ __launch_bounds__(256) void k_agg(
    const float* __restrict__ x, const _Float16* __restrict__ he,
    const float* __restrict__ atto, const float* __restrict__ vmix,
    float* __restrict__ hagg, float* __restrict__ cnorm, float* __restrict__ dvo)
{
    __shared__ _Float16 s_he[NN * HH];    // 32 KB slab for this (b,i)
    __shared__ float s_u[NN][3];
    __shared__ float s_a[NN][4];
    __shared__ float red[256];

    int bi = blockIdx.x;
    int b  = bi >> 8;
    int i  = bi & 255;
    int t  = threadIdx.x;

    const _Float16* hsrc = he + (size_t)bi * NN * HH;

#if __has_builtin(__builtin_amdgcn_tensor_load_to_lds)
    // TDM descriptor: 1-D tile of 4096 x 8-byte elements (= 32 KB), no cluster
    // multicast, no padding/iteration. Issued by wave 0 only (TDM ignores EXEC).
    if (t < 32) {
        uint64_t gaddr   = (uint64_t)(uintptr_t)hsrc;
        uint32_t lds_off = (uint32_t)(uintptr_t)(&s_he[0]);
        u32x4 g0;
        g0[0] = 1u;                                   // count=1, user desc
        g0[1] = lds_off;                              // lds_addr
        g0[2] = (uint32_t)gaddr;                      // global_addr[31:0]
        g0[3] = (uint32_t)((gaddr >> 32) & 0x01FFFFFFu) | (2u << 30); // addr[56:32], type=2
        i32x8 g1;
        g1[0] = (int)(3u << 16);                      // wg_mask=0, data_size=8B
        g1[1] = (int)((4096u & 0xFFFFu) << 16);       // tensor_dim0[15:0]
        g1[2] = (int)((4096u >> 16) | (1u << 16));    // tensor_dim0[31:16] | tensor_dim1
        g1[3] = (int)(4096u << 16);                   // tile_dim0 = 4096
        g1[4] = 1;                                    // tile_dim1 = 1
        g1[5] = 4096;                                 // tensor_dim0_stride[31:0]
        g1[6] = 0;
        g1[7] = 0;
        i32x4 z4; z4[0] = 0; z4[1] = 0; z4[2] = 0; z4[3] = 0;
#if defined(__clang_major__) && (__clang_major__ >= 23)
        i32x8 z8; for (int q = 0; q < 8; ++q) z8[q] = 0;
        __builtin_amdgcn_tensor_load_to_lds(g0, g1, z4, z4, z8, 0);
#else
        __builtin_amdgcn_tensor_load_to_lds(g0, g1, z4, z4, 0);
#endif
        __builtin_amdgcn_s_wait_tensorcnt(0);
    }
#else
    // fallback: cooperative copy
    {
        const uint32_t* src32 = (const uint32_t*)hsrc;
        uint32_t* dst32 = (uint32_t*)&s_he[0];
        for (int idx = t; idx < NN * HH / 2; idx += 256) dst32[idx] = src32[idx];
    }
#endif

    // unit vectors + attention weights into LDS (thread t = j)
    {
        int j = t;
        float dx = x[(b * NN + j) * 3 + 0] - x[(b * NN + i) * 3 + 0];
        float dy = x[(b * NN + j) * 3 + 1] - x[(b * NN + i) * 3 + 1];
        float dz = x[(b * NN + j) * 3 + 2] - x[(b * NN + i) * 3 + 2];
        float dist = sqrtf(dx * dx + dy * dy + dz * dz + EPSV);
        float inv = 1.0f / (dist + EPSV);
        s_u[j][0] = dx * inv; s_u[j][1] = dy * inv; s_u[j][2] = dz * inv;
        const float* ap = &atto[((size_t)bi * NN + j) * 4];
        s_a[j][0] = ap[0]; s_a[j][1] = ap[1]; s_a[j][2] = ap[2]; s_a[j][3] = ap[3];
    }
    __syncthreads();

    int c = t, hh = c >> 2, head = c & 3;
    float acc = 0.0f, c0 = 0.0f, c1 = 0.0f, c2 = 0.0f;
    #pragma unroll 4
    for (int j = 0; j < NN; ++j) {
        float w = (float)s_he[j * HH + hh] * s_a[j][head];
        acc += w;
        c0 += w * s_u[j][0];
        c1 += w * s_u[j][1];
        c2 += w * s_u[j][2];
    }
    hagg[(size_t)bi * CC + c] = acc;
    const float inv_n = 1.0f / (float)NN;
    c0 *= inv_n; c1 *= inv_n; c2 *= inv_n;
    cnorm[(size_t)bi * CC + c] = c0 * c0 + c1 * c1 + c2 * c2;

    float vw = vmix[c];
    float d0 = blk_sum256(c0 * vw, red);
    float d1 = blk_sum256(c1 * vw, red);
    float d2 = blk_sum256(c2 * vw, red);
    if (t == 0) { dvo[bi * 3 + 0] = d0; dvo[bi * 3 + 1] = d1; dvo[bi * 3 + 2] = d2; }
}

// ---------------------------------------------------------------------------
// Kernel 5: post MLP, node MLP (+residual), velocity/coordinate update
// ---------------------------------------------------------------------------
__global__ __launch_bounds__(64) void k_node(
    const float* __restrict__ h, const float* __restrict__ x, const float* __restrict__ v,
    const float* __restrict__ hagg, const float* __restrict__ cnorm,
    const float* __restrict__ dvo,
    const float* __restrict__ pw1, const float* __restrict__ pb1,
    const float* __restrict__ pw2, const float* __restrict__ pb2,
    const float* __restrict__ nw1, const float* __restrict__ nb1,
    const float* __restrict__ nw2, const float* __restrict__ nb2,
    const float* __restrict__ vw1, const float* __restrict__ vb1,
    const float* __restrict__ vw2,
    float* __restrict__ out_h, float* __restrict__ out_x, float* __restrict__ out_v)
{
    __shared__ float s_cn[CC], s_ha[CC], s_h[FF], s_t0[HH], s_t1[HH], s_red[HH];
    int bi = blockIdx.x;
    int t  = threadIdx.x;

    #pragma unroll
    for (int q = 0; q < 4; ++q) {
        s_cn[t + 64 * q] = cnorm[(size_t)bi * CC + t + 64 * q];
        s_ha[t + 64 * q] = hagg [(size_t)bi * CC + t + 64 * q];
    }
    s_h[t] = h[(size_t)bi * FF + t];
    __syncthreads();

    // post MLP: comb_norm[256] -> 64 -> 64
    float a = pb1[t];
    for (int c = 0; c < CC; ++c) a += s_cn[c] * pw1[c * HH + t];
    s_t0[t] = silu_f(a);
    __syncthreads();
    float bsum = pb2[t];
    for (int k = 0; k < HH; ++k) bsum += s_t0[k] * pw2[k * HH + t];
    s_t1[t] = silu_f(bsum);        // h_comb
    __syncthreads();

    // node MLP: concat(h, h_agg, h_comb)[384] -> 64 -> 64, residual
    float n1 = nb1[t];
    for (int f = 0; f < FF; ++f) n1 += s_h[f]  * nw1[f * HH + t];
    for (int c = 0; c < CC; ++c) n1 += s_ha[c] * nw1[(FF + c) * HH + t];
    for (int k = 0; k < HH; ++k) n1 += s_t1[k] * nw1[(FF + CC + k) * HH + t];
    s_t0[t] = silu_f(n1);
    __syncthreads();
    float o = nb2[t];
    for (int k = 0; k < HH; ++k) o += s_t0[k] * nw2[k * FF + t];
    float hnew = s_h[t] + silu_f(o);
    out_h[(size_t)bi * FF + t] = hnew;
    s_t1[t] = hnew;
    __syncthreads();

    // velocity gate: v_scale = silu(h_new @ vw1 + vb1) @ vw2
    float vh = vb1[t];
    for (int k = 0; k < HH; ++k) vh += s_t1[k] * vw1[k * HH + t];
    s_red[t] = silu_f(vh) * vw2[t];
    __syncthreads();
    for (int s = 32; s > 0; s >>= 1) {
        if (t < s) s_red[t] += s_red[t + s];
        __syncthreads();
    }
    float vscale = s_red[0];
    if (t < 3) {
        float vn = dvo[bi * 3 + t] + vscale * v[(size_t)bi * 3 + t];
        out_v[(size_t)bi * 3 + t] = vn;
        out_x[(size_t)bi * 3 + t] = x[(size_t)bi * 3 + t] + vn;
    }
}

// ---------------------------------------------------------------------------
extern "C" void kernel_launch(void* const* d_in, const int* in_sizes, int n_in,
                              void* d_out, int out_size, void* d_ws, size_t ws_size,
                              hipStream_t stream)
{
    (void)in_sizes; (void)n_in; (void)out_size; (void)ws_size;
    const float* h    = (const float*)d_in[0];
    const float* x    = (const float*)d_in[1];
    const float* v    = (const float*)d_in[2];
    const float* ew1  = (const float*)d_in[3];
    const float* eb1  = (const float*)d_in[4];
    const float* ew2  = (const float*)d_in[5];
    const float* eb2  = (const float*)d_in[6];
    const float* semw = (const float*)d_in[7];
    const float* semb = (const float*)d_in[8];
    const float* pw1  = (const float*)d_in[9];
    const float* pb1  = (const float*)d_in[10];
    const float* pw2  = (const float*)d_in[11];
    const float* pb2  = (const float*)d_in[12];
    const float* nw1  = (const float*)d_in[13];
    const float* nb1  = (const float*)d_in[14];
    const float* nw2  = (const float*)d_in[15];
    const float* nb2  = (const float*)d_in[16];
    const float* vw1  = (const float*)d_in[17];
    const float* vb1  = (const float*)d_in[18];
    const float* vw2  = (const float*)d_in[19];
    const float* vmix = (const float*)d_in[20];
    const float* lg   = (const float*)d_in[21];

    const int BN = BATCH * NN;   // 512
    float* ws = (float*)d_ws;
    float* Anode = ws;  ws += (size_t)BN * HH;
    float* Bnode = ws;  ws += (size_t)BN * HH;
    float* semo  = ws;  ws += (size_t)BN * NN * 4;
    float* atto  = ws;  ws += (size_t)BN * NN * 4;
    float* hagg  = ws;  ws += (size_t)BN * CC;
    float* cnorm = ws;  ws += (size_t)BN * CC;
    float* dvo   = ws;  ws += (size_t)BN * 3;
    // 16-byte align the f16 slab for the 8-byte-element TDM descriptor
    _Float16* he = (_Float16*)(((uintptr_t)ws + 15) & ~(uintptr_t)15);

    float* out_h = (float*)d_out;
    float* out_x = out_h + (size_t)BN * FF;
    float* out_v = out_x + (size_t)BN * 3;

    k_prep<<<BN, 64, 0, stream>>>(h, ew1, eb1, Anode, Bnode);
    k_edge<<<BN, 256, 0, stream>>>(x, Anode, Bnode, ew1, ew2, eb2, semw, semb, he, semo);
    k_att <<<BN, 256, 0, stream>>>(x, semo, lg, atto);
    k_agg <<<BN, 256, 0, stream>>>(x, he, atto, vmix, hagg, cnorm, dvo);
    k_node<<<BN, 64, 0, stream>>>(h, x, v, hagg, cnorm, dvo,
                                  pw1, pb1, pw2, pb2, nw1, nb1, nw2, nb2,
                                  vw1, vb1, vw2, out_h, out_x, out_v);
}
